// SimpleHeadAttention_26276609917218
// MI455X (gfx1250) — matrix operand order repaired and checked by reading, not compiled, not run on hardware
//
#include <hip/hip_runtime.h>

typedef __attribute__((ext_vector_type(16))) _Float16 v16h;
typedef __attribute__((ext_vector_type(8)))  _Float16 v8h;
typedef __attribute__((ext_vector_type(8)))  float    v8f;

#define B_ 2
#define N_ 2048
#define C_ 1024
#define H_ 16
#define D_ 64
#define SCALE_ 0.125f
#define NEG_ (-1e9f)

__device__ __forceinline__ v8f wmma_f16(v16h a, v16h b, v8f c) {
  // D = A(16x32 f16) * B(32x16 f16) + C(16x16 f32)
  return __builtin_amdgcn_wmma_f32_16x16x32_f16(false, a, false, b, (short)0, c,
                                                false, false);
}

// ---------------------------------------------------------------------------
// Kernel 1: qkv = x @ qkv_w^T + qkv_b ; scatter to q (scaled, f16 [B,H,N,D]),
// k (f16 [B,H,N,D]), vT (f16 [B,H,D,N]).
// M = B*N = 4096, Nd = 3C = 3072, K = C = 1024.
// Each wave computes a 16x64 output strip (4 tiles share one A-fragment).
// ---------------------------------------------------------------------------
__global__ __launch_bounds__(256) void qkv_gemm_kernel(
    const float* __restrict__ x, const float* __restrict__ qkv_w,
    const float* __restrict__ qkv_b, _Float16* __restrict__ q,
    _Float16* __restrict__ k, _Float16* __restrict__ vT) {
  const int wave = threadIdx.x >> 5;
  const int lane = threadIdx.x & 31;
  const int l = lane & 15;
  const int hi = lane >> 4;

  const int NG = (3 * C_) / 64;  // 48 column groups of 64
  const int wg = blockIdx.x * 8 + wave;
  const int mt = wg / NG;
  const int ng = wg % NG;
  const int i0 = mt * 16;
  const int j0 = ng * 64;

  const float* xrow = x + (size_t)(i0 + l) * C_;
  const float* wrow = qkv_w + (size_t)(j0 + l) * C_;  // tile t adds t*16*C_

  v8f acc[4] = {};
  for (int kb = 0; kb < C_; kb += 32) {
    v16h a;
#pragma unroll
    for (int e = 0; e < 16; e += 2) {
      // A layout: k = g*16 + hi*8 + o
      const int ka = kb + (e >> 3) * 16 + hi * 8 + (e & 7);
      const float2 xv = *(const float2*)(xrow + ka);
      a[e] = (_Float16)xv.x;
      a[e + 1] = (_Float16)xv.y;
    }
#pragma unroll
    for (int t = 0; t < 4; ++t) {
      v16h b;
      const float* wr = wrow + (size_t)t * 16 * C_;
#pragma unroll
      for (int e = 0; e < 16; e += 2) {
        // B layout: k = g*16 + hi*8 + o ; B[k][n] = W[j0+t*16+n][kb+k]
        const float2 wv = *(const float2*)(wr + kb + (e >> 3) * 16 + hi * 8 + (e & 7));
        b[e] = (_Float16)wv.x;
        b[e + 1] = (_Float16)wv.y;
      }
      acc[t] = wmma_f16(a, b, acc[t]);
    }
  }

  const int bb = i0 / N_;        // batch (row stripe never crosses batches)
  const int nrow = i0 % N_;      // base sequence position
#pragma unroll
  for (int t = 0; t < 4; ++t) {
    const int jt0 = j0 + t * 16;       // wave-uniform column base
    const int region = jt0 >> 10;      // 0=q, 1=k, 2=v (uniform)
    const int jl = jt0 + l;
    const float bias = qkv_b[jl];
    const int jj = jl & (C_ - 1);
    const int hh = jj >> 6;
    const int d = jj & 63;
    if (region == 0) {
      _Float16* dst = q + ((size_t)(bb * H_ + hh) * N_ + nrow) * D_ + d;
#pragma unroll
      for (int r = 0; r < 8; ++r)
        dst[(size_t)(r + hi * 8) * D_] =
            (_Float16)((acc[t][r] + bias) * SCALE_);
    } else if (region == 1) {
      _Float16* dst = k + ((size_t)(bb * H_ + hh) * N_ + nrow) * D_ + d;
#pragma unroll
      for (int r = 0; r < 8; ++r)
        dst[(size_t)(r + hi * 8) * D_] = (_Float16)(acc[t][r] + bias);
    } else {
      _Float16* dst = vT + ((size_t)(bb * H_ + hh) * D_ + d) * N_ + nrow;
#pragma unroll
      for (int r = 0; r < 8; ++r)
        dst[r + hi * 8] = (_Float16)(acc[t][r] + bias);
    }
  }
}

// ---------------------------------------------------------------------------
// Kernel 2: fused attention. One workgroup = (b, h, 16-row stripe).
// S stripe (16 x 2048 f32) lives in LDS; softmax in LDS; attn written once;
// normalized probs also stored as f16 overlaying the same LDS rows, so the
// PV phase feeds WMMA A-fragments with two 16B ds reads per fragment.
// ---------------------------------------------------------------------------
__global__ __launch_bounds__(256) void attn_kernel(
    const _Float16* __restrict__ q, const _Float16* __restrict__ kmat,
    const _Float16* __restrict__ vT, const float* __restrict__ int_matrix,
    const float* __restrict__ mask, float* __restrict__ attn_out,
    _Float16* __restrict__ head_out) {
  constexpr int SP = 2052;               // padded row stride (bank-conflict dodge)
  __shared__ float S[16][SP];            // 131,328 B
  __shared__ float Obuf[2][16][64];      // 8,192 B

  const int wave = threadIdx.x >> 5;
  const int lane = threadIdx.x & 31;
  const int l = lane & 15;
  const int hi = lane >> 4;

  int bt = blockIdx.x;
  const int it = bt & 127;  bt >>= 7;
  const int h = bt & 15;    bt >>= 4;
  const int b = bt;
  const int i0 = it * 16;

  const _Float16* qbase = q + (size_t)(b * H_ + h) * N_ * D_;
  const _Float16* kbase = kmat + (size_t)(b * H_ + h) * N_ * D_;
  const _Float16* vbase = vT + (size_t)(b * H_ + h) * D_ * N_;
  const float* imbase = int_matrix + (size_t)(b * H_ + h) * N_ * N_;
  const float* mbase = mask + (size_t)b * N_ * N_;
  float* abase = attn_out + (size_t)(b * H_ + h) * N_ * N_;

  // ---- load Q A-fragments (16x64 -> two 16x32 fragments), already scaled ----
  v16h aq0, aq1;
  {
    const _Float16* qr = qbase + (size_t)(i0 + l) * D_;
    v8h t0 = *(const v8h*)(qr + hi * 16);
    v8h t1 = *(const v8h*)(qr + hi * 16 + 8);
    v8h t2 = *(const v8h*)(qr + 32 + hi * 16);
    v8h t3 = *(const v8h*)(qr + 40 + hi * 16);
#pragma unroll
    for (int e = 0; e < 8; ++e) {
      aq0[e] = t0[e]; aq0[e + 8] = t1[e];
      aq1[e] = t2[e]; aq1[e + 8] = t3[e];
    }
  }

  // ---- phase 1: S = Q K^T + bias, into LDS. Each wave covers 256 columns ----
#pragma unroll 1
  for (int t = 0; t < 16; ++t) {
    const int j0 = (wave * 16 + t) * 16;
    const _Float16* kr = kbase + (size_t)(j0 + l) * D_;
    v16h b0 = *(const v16h*)(kr + hi * 16);
    v16h b1 = *(const v16h*)(kr + 32 + hi * 16);
    v8f acc = {};
    acc = wmma_f16(aq0, b0, acc);
    acc = wmma_f16(aq1, b1, acc);
#pragma unroll
    for (int r = 0; r < 8; ++r) {
      const int m = r + hi * 8;
      const int i = i0 + m;
      const int j = j0 + l;
      const float bias =
          imbase[(size_t)i * N_ + j] + (1.0f - mbase[(size_t)i * N_ + j]) * NEG_;
      S[m][j] = acc[r] + bias;
    }
  }
  __syncthreads();

  // ---- phase 2: softmax per row (2 rows per wave), write attn once.
  // Normalized p is also stored as f16 overlaying the row's own storage:
  // the f32 load of index c (earlier instruction) precedes the f16 store
  // covering float index c/2 <= c; per-wave LDS ordering makes this safe. ----
#pragma unroll 1
  for (int rr = 0; rr < 2; ++rr) {
    const int m = wave * 2 + rr;
    float* row = S[m];
    _Float16* rowh = (_Float16*)row;     // f16 overlay of the same row
    float vmax = -3.4e38f;
    for (int c = lane; c < N_; c += 32) vmax = fmaxf(vmax, row[c]);
#pragma unroll
    for (int off = 16; off >= 1; off >>= 1)
      vmax = fmaxf(vmax, __shfl_xor(vmax, off, 32));
    float vsum = 0.0f;
    for (int c = lane; c < N_; c += 32) {
      const float e = __expf(row[c] - vmax);
      row[c] = e;
      vsum += e;
    }
#pragma unroll
    for (int off = 16; off >= 1; off >>= 1) vsum += __shfl_xor(vsum, off, 32);
    const float inv = 1.0f / vsum;
    float* arow = abase + (size_t)(i0 + m) * N_;
    for (int c = lane; c < N_; c += 32) {
      const float p = row[c] * inv;
      arow[c] = p;                 // coalesced global write, exactly once
      rowh[c] = (_Float16)p;       // f16 overlay for the PV phase
    }
  }
  __syncthreads();

  // ---- phase 3: O = P @ V  (16x64). wave -> (d-tile = w&3, K-half = w>>2) ----
  {
    const int t = wave & 3;
    const int kh = wave >> 2;
    const _Float16* rowh = (const _Float16*)&S[l][0];          // P row l, f16
    const _Float16* vrow = vbase + (size_t)(t * 16 + l) * N_;  // vT row: fixed d
    v8f o = {};
    const int jend = kh * 1024 + 1024;
#pragma unroll 1
    for (int jb = kh * 1024; jb < jend; jb += 32) {
      v16h ap;
      const v8h p0 = *(const v8h*)(rowh + jb + hi * 16);       // k = hi*16+0..7
      const v8h p1 = *(const v8h*)(rowh + jb + hi * 16 + 8);   // k = hi*16+8+..
#pragma unroll
      for (int e = 0; e < 8; ++e) {
        ap[e] = p0[e];
        ap[e + 8] = p1[e];
      }
      const v16h bv = *(const v16h*)(vrow + jb + hi * 16);
      o = wmma_f16(ap, bv, o);
    }
#pragma unroll
    for (int r = 0; r < 8; ++r) {
      const int m = r + hi * 8;
      Obuf[kh][m][t * 16 + l] = o[r];
    }
  }
  __syncthreads();

  // ---- combine K-halves, emit f16 head output [B,N,C] layout ----
  for (int e = threadIdx.x; e < 16 * 64; e += 256) {
    const int m = e >> 6;
    const int d = e & 63;
    const float val = Obuf[0][m][d] + Obuf[1][m][d];
    head_out[((size_t)(b * N_ + i0 + m)) * C_ + h * D_ + d] = (_Float16)val;
  }
}

// ---------------------------------------------------------------------------
// Kernel 3: out = head_out(f16) @ proj_w^T + proj_b (4096 x 1024 x 1024).
// Each wave computes a 16x64 strip (4 tiles share one A-fragment).
// ---------------------------------------------------------------------------
__global__ __launch_bounds__(256) void proj_gemm_kernel(
    const _Float16* __restrict__ a, const float* __restrict__ pw,
    const float* __restrict__ pb, float* __restrict__ out) {
  const int wave = threadIdx.x >> 5;
  const int lane = threadIdx.x & 31;
  const int l = lane & 15;
  const int hi = lane >> 4;

  const int NG = C_ / 64;  // 16 column groups
  const int wg = blockIdx.x * 8 + wave;
  const int mt = wg / NG;
  const int ng = wg % NG;
  const int i0 = mt * 16;
  const int j0 = ng * 64;

  const _Float16* ar = a + (size_t)(i0 + l) * C_;
  const float* wrow = pw + (size_t)(j0 + l) * C_;

  v8f acc[4] = {};
  for (int kb = 0; kb < C_; kb += 32) {
    v16h av;
    const v8h t0 = *(const v8h*)(ar + kb + hi * 8);
    const v8h t1 = *(const v8h*)(ar + kb + 16 + hi * 8);
#pragma unroll
    for (int e = 0; e < 8; ++e) {
      av[e] = t0[e];
      av[e + 8] = t1[e];
    }
#pragma unroll
    for (int t = 0; t < 4; ++t) {
      v16h bv;
      const float* wr = wrow + (size_t)t * 16 * C_;
#pragma unroll
      for (int e = 0; e < 16; e += 2) {
        const float2 wv = *(const float2*)(wr + kb + (e >> 3) * 16 + hi * 8 + (e & 7));
        bv[e] = (_Float16)wv.x;
        bv[e + 1] = (_Float16)wv.y;
      }
      acc[t] = wmma_f16(av, bv, acc[t]);
    }
  }

#pragma unroll
  for (int t = 0; t < 4; ++t) {
    const int j = j0 + t * 16 + l;
    const float bias = pb[j];
#pragma unroll
    for (int r = 0; r < 8; ++r) {
      const int i = i0 + r + hi * 8;
      out[(size_t)i * C_ + j] = acc[t][r] + bias;
    }
  }
}

// ---------------------------------------------------------------------------
extern "C" void kernel_launch(void* const* d_in, const int* in_sizes, int n_in,
                              void* d_out, int out_size, void* d_ws,
                              size_t ws_size, hipStream_t stream) {
  const float* x = (const float*)d_in[0];
  const float* int_matrix = (const float*)d_in[1];
  const float* mask = (const float*)d_in[2];
  const float* qkv_w = (const float*)d_in[3];
  const float* qkv_b = (const float*)d_in[4];
  const float* proj_w = (const float*)d_in[5];
  const float* proj_b = (const float*)d_in[6];

  float* out = (float*)d_out;                          // [B,N,C]
  float* attn = out + (size_t)B_ * N_ * C_;            // [B,H,N,N]

  _Float16* wsh = (_Float16*)d_ws;
  const size_t QL = (size_t)B_ * H_ * N_ * D_;  // 4,194,304 halves each
  _Float16* q = wsh;
  _Float16* k = wsh + QL;
  _Float16* vT = wsh + 2 * QL;
  _Float16* hout = wsh + 3 * QL;  // [B,N,C] f16, 32 MB ws total

  // 1) QKV projection: 256 row tiles x 48 column groups, 8 waves per block
  qkv_gemm_kernel<<<(256 * 48) / 8, 256, 0, stream>>>(x, qkv_w, qkv_b, q, k,
                                                      vT);
  // 2) fused attention: B*H*(N/16) workgroups
  attn_kernel<<<B_ * H_ * (N_ / 16), 256, 0, stream>>>(q, k, vT, int_matrix,
                                                       mask, attn, hout);
  // 3) output projection: 256 row tiles x 16 column groups, 8 waves per block
  proj_gemm_kernel<<<(256 * 16) / 8, 256, 0, stream>>>(hout, proj_w, proj_b,
                                                       out);
}
